// CausalSelfAttention_56942676410843
// MI455X (gfx1250) — matrix operand order repaired and checked by reading, not compiled, not run on hardware
//
#include <hip/hip_runtime.h>
#include <cstdint>
#include <cstddef>

typedef __attribute__((ext_vector_type(16))) __bf16 v16bf;
typedef __attribute__((ext_vector_type(4)))  __bf16 v4bf;
typedef __attribute__((ext_vector_type(8)))  float  v8f;
typedef __attribute__((ext_vector_type(4)))  unsigned int v4u;
typedef __attribute__((ext_vector_type(8)))  int v8i;
typedef __attribute__((ext_vector_type(4)))  int v4i;

constexpr int kB  = 4, kT = 2048, kC = 1024, kH = 16, kHD = 64;
constexpr int kBT = kB * kT;   // 8192 rows (b*T+t)
constexpr int k3C = 3 * kC;    // 3072

__device__ __forceinline__ v8f wmma_bf16(v16bf a, v16bf b, v8f c) {
  // D = A(16x32 bf16) * B(32x16 bf16) + C(16x16 f32)
  return __builtin_amdgcn_wmma_f32_16x16x32_bf16(
      /*neg_a=*/false, a, /*neg_b=*/false, b,
      /*c_mod=*/(short)0, c, /*reuse_a=*/false, /*reuse_b=*/false);
}

// LDS byte offset of a __shared__ pointer: generic shared address keeps the
// LDS offset in the low 32 bits (aperture in the high bits).
__device__ __forceinline__ unsigned int lds_off(const void* p) {
  return (unsigned int)(uintptr_t)p;
}

// Issue one TDM 2-D tile copy global->LDS (bf16 elements).
// D# packing per CDNA5 ISA:
//   group0: count=1 | lds_addr | global_addr[56:0] | type=2
//   group1: data_size=1(2B), tensor_dim0/1 large (no clip), tile_w x tile_h,
//           tensor_dim0_stride = row stride (elements)
__device__ __forceinline__ void tdm_load_2d(unsigned int lds_byte_off,
                                            const void* gptr,
                                            unsigned int tile_w,
                                            unsigned int tile_h,
                                            unsigned long long stride_elems) {
  const unsigned long long ga = (unsigned long long)(uintptr_t)gptr;
  v4u g0;
  g0.x = 1u;                                        // count=1 (valid user D#)
  g0.y = lds_byte_off;                              // lds_addr
  g0.z = (unsigned int)(ga & 0xFFFFFFFFu);          // global_addr[31:0]
  g0.w = (unsigned int)((ga >> 32) & 0x1FFFFFFu)    // global_addr[56:32]
         | (2u << 30);                              // type = 2 ("image")
  const unsigned int dim0 = 1u << 24;               // generous bounds: no clip
  const unsigned int dim1 = 1u << 24;
  v8i g1;
  g1[0] = (int)(1u << 16);                          // data_size=1 -> 2 bytes
  g1[1] = (int)((dim0 & 0xFFFFu) << 16);            // tensor_dim0[15:0]
  g1[2] = (int)((dim0 >> 16) | ((dim1 & 0xFFFFu) << 16));
  g1[3] = (int)((dim1 >> 16) | (tile_w << 16));     // tile_dim0
  g1[4] = (int)(tile_h);                            // tile_dim1 (tile_dim2=0)
  g1[5] = (int)(unsigned int)(stride_elems & 0xFFFFFFFFu);
  g1[6] = (int)(unsigned int)(stride_elems >> 32);  // tensor_dim1_stride=0
  g1[7] = 0;
  v4i z4 = {0, 0, 0, 0};
  v8i z8 = {0, 0, 0, 0, 0, 0, 0, 0};
  // clang-23 / therock-10.0 headers: 6-arg form (g0, g1, g2, g3, ext, cpol)
  __builtin_amdgcn_tensor_load_to_lds(g0, g1, z4, z4, z8, /*cpol=*/0);
}

// ---------------------------------------------------------------- conversion
__global__ void cvt_f32_to_bf16(const float* __restrict__ src,
                                __bf16* __restrict__ dst, int n) {
  int i = (blockIdx.x * blockDim.x + threadIdx.x) * 4;
  if (i + 3 < n) {
    const float4 f = *reinterpret_cast<const float4*>(src + i);
    v4bf h = { (__bf16)f.x, (__bf16)f.y, (__bf16)f.z, (__bf16)f.w };
    *reinterpret_cast<v4bf*>(dst + i) = h;
  } else {
    for (; i < n; ++i) dst[i] = (__bf16)src[i];
  }
}

// --------------------------------------------------- TDM-staged GEMM body
// Block = 256 threads = 8 waves as 2 (row) x 4 (col); block tile 64 x 128.
// Each wave computes a 32x32 output (2x2 WMMA tiles). K stepped by 32 with
// double-buffered TDM staging: A tile 64x32 bf16 (4KB), B tile 128x32 (8KB).
struct GemmAcc { v8f a[2][2]; };

__device__ __forceinline__ GemmAcc
gemm_lds_body(const __bf16* __restrict__ A, const __bf16* __restrict__ B,
              int row0, int col0,
              __bf16 (*As)[64 * 32], __bf16 (*Bs)[128 * 32]) {
  const int lane  = threadIdx.x & 31;
  const int l16   = lane & 15;
  const int lhalf = lane >> 4;
  const int wid   = threadIdx.x >> 5;
  const int wrow  = wid & 1;
  const int wcol  = wid >> 1;
  const bool issuer = (threadIdx.x < 32);

  GemmAcc acc = {};
  if (issuer) {
    tdm_load_2d(lds_off(&As[0][0]), A + (size_t)row0 * kC, 32, 64,  kC);
    tdm_load_2d(lds_off(&Bs[0][0]), B + (size_t)col0 * kC, 32, 128, kC);
  }
  int buf = 0;
  for (int k0 = 0; k0 < kC; k0 += 32) {
    if (issuer) {
      if (k0 + 32 < kC) {
        tdm_load_2d(lds_off(&As[buf ^ 1][0]),
                    A + (size_t)row0 * kC + k0 + 32, 32, 64,  kC);
        tdm_load_2d(lds_off(&Bs[buf ^ 1][0]),
                    B + (size_t)col0 * kC + k0 + 32, 32, 128, kC);
        __builtin_amdgcn_s_wait_tensorcnt(2);  // current buf's (older) pair done
      } else {
        __builtin_amdgcn_s_wait_tensorcnt(0);
      }
    }
    __syncthreads();                            // buf ready for everyone
    const __bf16* a = &As[buf][wrow * 32 * 32];
    const __bf16* b = &Bs[buf][wcol * 32 * 32];
    v16bf af[2], bf[2];
#pragma unroll
    for (int i = 0; i < 2; ++i) {
      af[i] = *(const v16bf*)(a + (i * 16 + l16) * 32 + lhalf * 16);
      bf[i] = *(const v16bf*)(b + (i * 16 + l16) * 32 + lhalf * 16);
    }
#pragma unroll
    for (int mt = 0; mt < 2; ++mt)
#pragma unroll
      for (int nt = 0; nt < 2; ++nt)
        acc.a[mt][nt] = wmma_bf16(af[mt], bf[nt], acc.a[mt][nt]);
    __syncthreads();                            // done reading buf; reuse ok
    buf ^= 1;
  }
  return acc;
}

// ------------------------------------------------------------------ qkv GEMM
// qkv[row, o] = sum_c x[row, c] * w_attn[o, c] + b_attn[o]
__global__ void __launch_bounds__(256)
qkv_gemm(const __bf16* __restrict__ X, const __bf16* __restrict__ W,
         const float* __restrict__ bias,
         __bf16* __restrict__ Q, __bf16* __restrict__ K, __bf16* __restrict__ V) {
  __shared__ __bf16 As[2][64 * 32];
  __shared__ __bf16 Bs[2][128 * 32];
  const int lane  = threadIdx.x & 31;
  const int l16   = lane & 15;
  const int lhalf = lane >> 4;
  const int wid   = threadIdx.x >> 5;
  const int wrow  = wid & 1;
  const int wcol  = wid >> 1;
  const int row0  = blockIdx.x * 64;
  const int col0  = blockIdx.y * 128;

  GemmAcc acc = gemm_lds_body(X, W, row0, col0, As, Bs);

#pragma unroll
  for (int nt = 0; nt < 2; ++nt) {
    const int o     = col0 + wcol * 32 + nt * 16 + l16;
    const float bv  = bias[o];
    const int which = o >> 10;
    const int rem   = o & (kC - 1);
    const int h     = rem >> 6;
    const int d     = rem & (kHD - 1);
    __bf16* dst = (which == 0) ? Q : (which == 1) ? K : V;
#pragma unroll
    for (int mt = 0; mt < 2; ++mt) {
#pragma unroll
      for (int r = 0; r < 8; ++r) {
        const int row = row0 + wrow * 32 + mt * 16 + lhalf * 8 + r;
        const int bb  = row >> 11;
        const int t   = row & (kT - 1);
        dst[(((size_t)bb * kH + h) * kT + t) * kHD + d] =
            (__bf16)(acc.a[mt][nt][r] + bv);
      }
    }
  }
}

// ----------------------------------------------------------------- projection
__global__ void __launch_bounds__(256)
proj_gemm(const __bf16* __restrict__ Yb, const __bf16* __restrict__ W,
          const float* __restrict__ bias, float* __restrict__ Out) {
  __shared__ __bf16 As[2][64 * 32];
  __shared__ __bf16 Bs[2][128 * 32];
  const int lane  = threadIdx.x & 31;
  const int l16   = lane & 15;
  const int lhalf = lane >> 4;
  const int wid   = threadIdx.x >> 5;
  const int wrow  = wid & 1;
  const int wcol  = wid >> 1;
  const int row0  = blockIdx.x * 64;
  const int col0  = blockIdx.y * 128;

  GemmAcc acc = gemm_lds_body(Yb, W, row0, col0, As, Bs);

#pragma unroll
  for (int nt = 0; nt < 2; ++nt) {
    const int o    = col0 + wcol * 32 + nt * 16 + l16;
    const float bv = bias[o];
#pragma unroll
    for (int mt = 0; mt < 2; ++mt) {
#pragma unroll
      for (int r = 0; r < 8; ++r) {
        const int row = row0 + wrow * 32 + mt * 16 + lhalf * 8 + r;
        Out[(size_t)row * kC + o] = acc.a[mt][nt][r] + bv;
      }
    }
  }
}

// ------------------------------------------------------------- flash attention
// One wave owns 16 q rows; online softmax; P transposed via per-wave LDS.
__global__ void __launch_bounds__(128)
flash_attn(const __bf16* __restrict__ Q, const __bf16* __restrict__ K,
           const __bf16* __restrict__ V, __bf16* __restrict__ Y) {
  __shared__ __bf16 lds_p[4][16 * 32];        // per-wave 16x32 bf16 P tile
  const int lane  = threadIdx.x & 31;
  const int wid   = threadIdx.x >> 5;
  const int l16   = lane & 15;
  const int lhalf = lane >> 4;
  const int bh    = blockIdx.y;                // b*H + h
  const int b     = bh >> 4;
  const int h     = bh & 15;
  const int q0    = blockIdx.x * 64 + wid * 16;

  const __bf16* Qp = Q + (size_t)bh * kT * kHD;
  const __bf16* Kp = K + (size_t)bh * kT * kHD;
  const __bf16* Vp = V + (size_t)bh * kT * kHD;

  const v16bf qa0 = *(const v16bf*)(Qp + (size_t)(q0 + l16) * kHD + lhalf * 16);
  const v16bf qa1 = *(const v16bf*)(Qp + (size_t)(q0 + l16) * kHD + 32 + lhalf * 16);

  v8f oacc[4] = { v8f{}, v8f{}, v8f{}, v8f{} };
  float mrun[8], lrun[8];
#pragma unroll
  for (int r = 0; r < 8; ++r) { mrun[r] = -1e30f; lrun[r] = 0.f; }
  const float scale = 0.125f;                  // 1/sqrt(64)
  __bf16* pbuf = lds_p[wid];

  for (int kb = 0; kb <= q0 + 15; kb += 32) {  // causal: skip fully-masked blocks
    v8f s0 = {}, s1 = {};
    {
      v16bf k00 = *(const v16bf*)(Kp + (size_t)(kb + l16) * kHD + lhalf * 16);
      v16bf k01 = *(const v16bf*)(Kp + (size_t)(kb + l16) * kHD + 32 + lhalf * 16);
      s0 = wmma_bf16(qa0, k00, s0);
      s0 = wmma_bf16(qa1, k01, s0);
      v16bf k10 = *(const v16bf*)(Kp + (size_t)(kb + 16 + l16) * kHD + lhalf * 16);
      v16bf k11 = *(const v16bf*)(Kp + (size_t)(kb + 16 + l16) * kHD + 32 + lhalf * 16);
      s1 = wmma_bf16(qa0, k10, s1);
      s1 = wmma_bf16(qa1, k11, s1);
    }
#pragma unroll
    for (int r = 0; r < 8; ++r) {
      const int m  = lhalf * 8 + r;
      const int qg = q0 + m;
      float e0 = s0[r] * scale;
      float e1 = s1[r] * scale;
      if (kb + l16      > qg) e0 = -1e30f;
      if (kb + 16 + l16 > qg) e1 = -1e30f;
      float mx = fmaxf(e0, e1);
#pragma unroll
      for (int off = 1; off < 16; off <<= 1)
        mx = fmaxf(mx, __shfl_xor(mx, off, 32));
      const float mnew = fmaxf(mrun[r], mx);
      const float corr = __expf(mrun[r] - mnew);
      const float p0 = __expf(e0 - mnew);
      const float p1 = __expf(e1 - mnew);
      float rs = p0 + p1;
#pragma unroll
      for (int off = 1; off < 16; off <<= 1)
        rs += __shfl_xor(rs, off, 32);
      lrun[r] = lrun[r] * corr + rs;
      mrun[r] = mnew;
#pragma unroll
      for (int dt = 0; dt < 4; ++dt) oacc[dt][r] *= corr;
      pbuf[m * 32 + l16]      = (__bf16)p0;   // transpose staging (C->A layout)
      pbuf[m * 32 + 16 + l16] = (__bf16)p1;
    }
    asm volatile("" ::: "memory");   // DS in-order per wave; block reordering
    const v16bf pa = *(const v16bf*)(pbuf + l16 * 32 + lhalf * 16);
    asm volatile("" ::: "memory");
#pragma unroll
    for (int dt = 0; dt < 4; ++dt) {
      v16bf vb;
#pragma unroll
      for (int s = 0; s < 16; ++s)
        vb[s] = Vp[(size_t)(kb + lhalf * 16 + s) * kHD + dt * 16 + l16];
      oacc[dt] = wmma_bf16(pa, vb, oacc[dt]);
    }
  }

#pragma unroll
  for (int r = 0; r < 8; ++r) {
    const float inv = 1.0f / lrun[r];
    const int t = q0 + lhalf * 8 + r;
    __bf16* yrow = Y + ((size_t)b * kT + t) * kC + h * kHD;
#pragma unroll
    for (int dt = 0; dt < 4; ++dt)
      yrow[dt * 16 + l16] = (__bf16)(oacc[dt][r] * inv);
  }
}

// -------------------------------------------------------------------- launch
extern "C" void kernel_launch(void* const* d_in, const int* in_sizes, int n_in,
                              void* d_out, int out_size, void* d_ws, size_t ws_size,
                              hipStream_t stream) {
  (void)in_sizes; (void)n_in; (void)out_size; (void)ws_size;
  const float* x      = (const float*)d_in[0];
  const float* w_attn = (const float*)d_in[1];
  const float* b_attn = (const float*)d_in[2];
  const float* w_proj = (const float*)d_in[3];
  const float* b_proj = (const float*)d_in[4];
  float* out = (float*)d_out;

  char* ws = (char*)d_ws;
  size_t off = 0;
  auto alloc = [&](size_t bytes) -> char* {
    char* p = ws + off;
    off += (bytes + 255) & ~size_t(255);
    return p;
  };
  __bf16* xb  = (__bf16*)alloc((size_t)kBT * kC * 2);   // x in bf16
  __bf16* wab = (__bf16*)alloc((size_t)k3C * kC * 2);   // w_attn bf16
  __bf16* wpb = (__bf16*)alloc((size_t)kC  * kC * 2);   // w_proj bf16
  __bf16* qb  = (__bf16*)alloc((size_t)kBT * kC * 2);   // q [B,H,T,hd]
  __bf16* kb  = (__bf16*)alloc((size_t)kBT * kC * 2);   // k [B,H,T,hd]
  __bf16* vb  = (__bf16*)alloc((size_t)kBT * kC * 2);   // v [B,H,T,hd]
  __bf16* yb  = (__bf16*)alloc((size_t)kBT * kC * 2);   // y [B,T,C]

  auto cvt = [&](const float* s, __bf16* d, int n) {
    const int threads = 256;
    const int elems = (n + 3) / 4;
    cvt_f32_to_bf16<<<(elems + threads - 1) / threads, threads, 0, stream>>>(s, d, n);
  };
  cvt(x,      xb,  kBT * kC);
  cvt(w_attn, wab, k3C * kC);
  cvt(w_proj, wpb, kC * kC);

  qkv_gemm <<<dim3(kBT / 64, k3C / 128), 256, 0, stream>>>(xb, wab, b_attn, qb, kb, vb);
  flash_attn<<<dim3(kT / 64, kB * kH),   128, 0, stream>>>(qb, kb, vb, yb);
  proj_gemm<<<dim3(kBT / 64, kC / 128),  256, 0, stream>>>(yb, wpb, b_proj, out);
}
